// LinearAttention_79628693668176
// MI455X (gfx1250) — matrix operand order repaired and checked by reading, not compiled, not run on hardware
//
#include <hip/hip_runtime.h>
#include <hip/hip_bf16.h>
#include <math.h>

// Problem constants (from reference)
#define B_     8
#define C_     256     // DIM
#define HID_   512     // HEADS*DIM_HEAD
#define OC3_   1536    // 3*HID
#define NPOS   4096    // 64*64
#define HEADS_ 8
#define DH_    64

typedef __attribute__((ext_vector_type(16))) _Float16 v16h;
typedef __attribute__((ext_vector_type(8)))  float    v8f;
typedef unsigned int v4u __attribute__((ext_vector_type(4)));
typedef int          v8i __attribute__((ext_vector_type(8)));
typedef int          v4i __attribute__((ext_vector_type(4)));

union FragU { v16h h; unsigned int u[8]; uint4 q4[2]; };

// ---------------------------------------------------------------------------
// WMMA fragment loaders (CDNA5 16-bit layouts, wave32).
// A (16x32, MxK), row-major [M][K]: lane holds M=lane&15; K pairs at
// offsets {0,2,4,6,16,18,20,22} + 8*(lane>=16)  -> dwords {0..3, 8..11}.
// B (32x16, KxN) loaded from B^T stored [N][K]: lane holds N=lane&15,
// K = k0 + 8*(lane>=16) + {0..7, 16..23}  -> two b128 loads (same K order as A).
// Works for both global and LDS-resident panels (addrspace inferred after
// inlining: global_load_b128 vs ds_load_b128).
// ---------------------------------------------------------------------------
__device__ __forceinline__ v16h load_a_frag(const _Float16* __restrict__ A,
                                            int lda, int m0, int k0, int lane) {
  const int m  = m0 + (lane & 15);
  const int kb = k0 + ((lane >> 4) << 3);
  const unsigned int* p = (const unsigned int*)(A + (size_t)m * lda + kb);
  FragU f;
  f.u[0] = p[0]; f.u[1] = p[1]; f.u[2] = p[2];  f.u[3] = p[3];
  f.u[4] = p[8]; f.u[5] = p[9]; f.u[6] = p[10]; f.u[7] = p[11];
  return f.h;
}

__device__ __forceinline__ v16h load_bt_frag(const _Float16* __restrict__ Bt,
                                             int ldb, int n0, int k0, int lane) {
  const int n  = n0 + (lane & 15);
  const int kb = k0 + ((lane >> 4) << 3);
  const uint4* p = (const uint4*)(Bt + (size_t)n * ldb + kb);
  FragU f;
  f.q4[0] = p[0]; f.q4[1] = p[2];
  return f.h;
}

__device__ __forceinline__ v8f wmma_f16(v16h a, v16h b, v8f c) {
  return __builtin_amdgcn_wmma_f32_16x16x32_f16(false, a, false, b,
                                                (short)0, c, false, false);
}

// ---------------------------------------------------------------------------
// Tensor Data Mover: DMA a 2-D tile (rows x cols halves, row stride in
// halves) from global memory into LDS.  D# layout per CDNA5 ISA.
// Issued by one wave per block; tracked on TENSORcnt.
// This toolchain exposes the 6-arg builtin:
//   (uint32x4 g0, int32x8 g1, int32x4, int32x4, int32x8, i32 cpol)
// ---------------------------------------------------------------------------
__device__ __forceinline__ unsigned lds_off_of(const void* p) {
  // Flat LDS addresses carry the LDS byte offset in addr[31:0] (ISA 10.2).
  return (unsigned)(unsigned long long)(size_t)p;
}

__device__ __forceinline__ void tdm_load_2d(unsigned lds_off, const void* gptr,
                                            unsigned rows, unsigned cols,
                                            unsigned row_stride) {
  const unsigned long long ga = (unsigned long long)(size_t)gptr;
  v4u g0;
  g0.x = 1u;                                                  // count=1 valid D#
  g0.y = lds_off;                                             // lds_addr (bytes)
  g0.z = (unsigned)ga;                                        // global_addr[31:0]
  g0.w = 0x80000000u | (unsigned)((ga >> 32) & 0x01FFFFFFu);  // addr[56:32] | type=2
  v8i g1;
  g1[0] = (int)(1u << 16);                                    // data_size=1 (2 bytes)
  g1[1] = (int)(cols << 16);                                  // tensor_dim0[15:0] @ [31:16]
  g1[2] = (int)((cols >> 16) & 0xFFFFu) | (int)(rows << 16);  // dim0 hi | tensor_dim1 lo
  g1[3] = (int)((rows >> 16) & 0xFFFFu) | (int)(cols << 16);  // dim1 hi | tile_dim0
  g1[4] = (int)rows;                                          // tile_dim1 ; tile_dim2=0
  g1[5] = (int)row_stride;                                    // tensor_dim0_stride[31:0]
  g1[6] = 0;                                                  // stride hi | dim1_stride lo
  g1[7] = 0;
  v4i g2 = {0, 0, 0, 0};                                      // 2-D tile: unused
  v4i g3 = {0, 0, 0, 0};
  v8i g4 = {0, 0, 0, 0, 0, 0, 0, 0};
  __builtin_amdgcn_tensor_load_to_lds(g0, g1, g2, g3, g4, 0);
}

// ---------------------------------------------------------------------------
// K1: per-(b,c) spatial mean / rstd over 4096 elements
// ---------------------------------------------------------------------------
__global__ void __launch_bounds__(256)
stats_kernel(const float* __restrict__ x, float* __restrict__ mean,
             float* __restrict__ rstd) {
  const int row = blockIdx.x;                 // b*C + c
  const float* p = x + (size_t)row * NPOS;
  float s = 0.f, s2 = 0.f;
  for (int i = threadIdx.x; i < NPOS; i += 256) {
    float v = p[i];
    s += v; s2 += v * v;
  }
  __shared__ float sh[256], sh2[256];
  sh[threadIdx.x] = s; sh2[threadIdx.x] = s2;
  __syncthreads();
  for (int off = 128; off > 0; off >>= 1) {
    if (threadIdx.x < off) {
      sh[threadIdx.x]  += sh[threadIdx.x + off];
      sh2[threadIdx.x] += sh2[threadIdx.x + off];
    }
    __syncthreads();
  }
  if (threadIdx.x == 0) {
    float mu  = sh[0] * (1.f / NPOS);
    float var = sh2[0] * (1.f / NPOS) - mu * mu;
    mean[row] = mu;
    rstd[row] = rsqrtf(var + 1e-5f);
  }
}

// ---------------------------------------------------------------------------
// K2: normalize + affine, write xn transposed [b][n][c] as f16 (B^T for GEMM1)
// ---------------------------------------------------------------------------
__global__ void __launch_bounds__(256)
norm_transpose_kernel(const float* __restrict__ x, const float* __restrict__ gamma,
                      const float* __restrict__ beta, const float* __restrict__ mean,
                      const float* __restrict__ rstd, _Float16* __restrict__ xnT) {
  const int row = blockIdx.x;                 // b*C + c
  const int b = row >> 8, c = row & 255;
  const float rs = rstd[row];
  const float g  = gamma[c] * rs;
  const float be = beta[c] - mean[row] * g;   // xn = x*g + be
  const float* p = x + (size_t)row * NPOS;
  _Float16* dst = xnT + (size_t)b * NPOS * C_ + c;
  for (int n = threadIdx.x; n < NPOS; n += 256)
    dst[(size_t)n * C_] = (_Float16)(p[n] * g + be);
}

// ---------------------------------------------------------------------------
// K2b: f32 -> f16 weight conversion
// ---------------------------------------------------------------------------
__global__ void __launch_bounds__(256)
f32_to_f16_kernel(const float* __restrict__ src, _Float16* __restrict__ dst, int count) {
  int i = blockIdx.x * 256 + threadIdx.x;
  if (i < count) dst[i] = (_Float16)src[i];
}

// ---------------------------------------------------------------------------
// K3: QKV GEMM  (M=1536, K=256, N=4096 per batch)  + fused rotary/ELU epilogue
// Block: 256 thr = 8 waves (4Mx2N), block tile 128x64, wave tile 32x32.
// B panel (64 x 256 halves = 32KB) staged in LDS once via TDM.
// Outputs: qh [bh][n][d], kT/vT [bh][d][n]   (f16)
// ---------------------------------------------------------------------------
__global__ void __launch_bounds__(256)
gemm_qkv_kernel(const _Float16* __restrict__ Wq, const _Float16* __restrict__ xnT,
                _Float16* __restrict__ qh, _Float16* __restrict__ kT,
                _Float16* __restrict__ vT) {
  __shared__ _Float16 ldsB[64 * C_];               // 32 KB
  const int b    = blockIdx.z;
  const int lane = threadIdx.x & 31, wave = threadIdx.x >> 5;
  const int wm = wave >> 1, wn = wave & 1;
  const int m_base = blockIdx.y * 128 + wm * 32;   // output channel o
  const int n_blk  = blockIdx.x * 64;              // position tile base
  const int n_base = n_blk + wn * 32;
  const _Float16* Bt = xnT + (size_t)b * NPOS * C_;

  if (wave == 0) {
    tdm_load_2d(lds_off_of(ldsB), Bt + (size_t)n_blk * C_, 64, C_, C_);
    __builtin_amdgcn_s_wait_tensorcnt(0);
  }
  __syncthreads();

  v8f acc[2][2] = {};
  for (int k0 = 0; k0 < C_; k0 += 32) {
    v16h a0 = load_a_frag(Wq, C_, m_base,      k0, lane);
    v16h a1 = load_a_frag(Wq, C_, m_base + 16, k0, lane);
    v16h b0 = load_bt_frag(ldsB, C_, wn * 32,      k0, lane);
    v16h b1 = load_bt_frag(ldsB, C_, wn * 32 + 16, k0, lane);
    acc[0][0] = wmma_f16(a0, b0, acc[0][0]);
    acc[0][1] = wmma_f16(a0, b1, acc[0][1]);
    acc[1][0] = wmma_f16(a1, b0, acc[1][0]);
    acc[1][1] = wmma_f16(a1, b1, acc[1][1]);
  }

  // Epilogue: rotary (pairs (2i,2i+1) are adjacent accumulator rows in the
  // same lane) then ELU for q/k; scatter to GEMM-ready layouts.
  const float L2_5_OVER_15 = 2.321928094887362f / 15.f;
  for (int ti = 0; ti < 2; ++ti) {
    for (int tj = 0; tj < 2; ++tj) {
      const int o_base = m_base + ti * 16 + ((lane >> 4) << 3);  // even
      const int n  = n_base + tj * 16 + (lane & 15);
      const int hr = n >> 6, wc = n & 63;
      for (int r = 0; r < 8; r += 2) {
        const int o = o_base + r;
        float v0 = acc[ti][tj][r], v1 = acc[ti][tj][r + 1];
        const int sel  = o >> 9;            // 0=q 1=k 2=v
        const int hid  = o & 511;
        const int head = hid >> 6, d = hid & 63;   // d even
        const size_t bh = (size_t)b * HEADS_ + head;
        if (sel < 2) {
          const int pidx = d >> 1;
          float pos, lsc;
          if (pidx < 16) { pos = -1.f + (2.f / 63.f) * hr; lsc = exp2f(pidx * L2_5_OVER_15); }
          else           { pos = -1.f + (2.f / 63.f) * wc; lsc = exp2f((pidx - 16) * L2_5_OVER_15); }
          const float th = pos * lsc * 3.14159265358979f;
          float sn, cs;
          __sincosf(th, &sn, &cs);
          float r0 = v0 * cs - v1 * sn;
          float r1 = v1 * cs + v0 * sn;
          if (sel == 0) {
            r0 = (r0 > 0.f ? r0 + 1.f : __expf(r0)) * 0.125f;   // (elu+1)*dh^-0.5
            r1 = (r1 > 0.f ? r1 + 1.f : __expf(r1)) * 0.125f;
            _Float16* q = qh + ((bh * NPOS + n) * DH_ + d);
            q[0] = (_Float16)r0; q[1] = (_Float16)r1;
          } else {
            r0 = (r0 > 0.f ? r0 + 1.f : __expf(r0));
            r1 = (r1 > 0.f ? r1 + 1.f : __expf(r1));
            _Float16* kk = kT + ((bh * DH_ + d) * NPOS + n);
            kk[0] = (_Float16)r0; kk[NPOS] = (_Float16)r1;
          }
        } else {
          _Float16* vv = vT + ((bh * DH_ + d) * NPOS + n);
          vv[0] = (_Float16)v0; vv[NPOS] = (_Float16)v1;
        }
      }
    }
  }
}

// ---------------------------------------------------------------------------
// K4: k_sum[bh][d] = sum_n kT[bh][d][n]
// ---------------------------------------------------------------------------
__global__ void __launch_bounds__(256)
ksum_kernel(const _Float16* __restrict__ kT, float* __restrict__ ksum) {
  const int row = blockIdx.x;                 // bh*64 + d
  const _Float16* p = kT + (size_t)row * NPOS;
  float s = 0.f;
  for (int i = threadIdx.x; i < NPOS; i += 256) s += (float)p[i];
  __shared__ float sh[256];
  sh[threadIdx.x] = s;
  __syncthreads();
  for (int off = 128; off > 0; off >>= 1) {
    if (threadIdx.x < off) sh[threadIdx.x] += sh[threadIdx.x + off];
    __syncthreads();
  }
  if (threadIdx.x == 0) ksum[row] = sh[0];
}

// ---------------------------------------------------------------------------
// K5: d_inv[bh][n] = 1 / (q[n,:] . k_sum)
// ---------------------------------------------------------------------------
__global__ void __launch_bounds__(256)
dinv_kernel(const _Float16* __restrict__ qh, const float* __restrict__ ksum,
            float* __restrict__ dinv) {
  const int idx = blockIdx.x * 256 + threadIdx.x;   // bh*4096 + n
  const int bh  = idx >> 12;
  const _Float16* q = qh + (size_t)idx * DH_;
  const float* ks = ksum + bh * DH_;
  float s = 0.f;
  for (int d = 0; d < DH_; ++d) s += (float)q[d] * ks[d];
  dinv[idx] = 1.f / s;
}

// ---------------------------------------------------------------------------
// K6: context GEMM per (b,h): ctx[d][e] = sum_n kT[d][n] * vT[e][n]
// M=64, N=64, K=4096.  8 waves (4Mx2N), wave tile 16x32.
// A and B panels (64 x 256 halves each) staged per K slab via TDM.
// Store transposed ctxT[bh][e][d] (f16) as B^T for K7.
// ---------------------------------------------------------------------------
__global__ void __launch_bounds__(256)
gemm_ctx_kernel(const _Float16* __restrict__ kT, const _Float16* __restrict__ vT,
                _Float16* __restrict__ ctxT) {
  __shared__ _Float16 ldsA[64 * 256];   // 32 KB
  __shared__ _Float16 ldsB[64 * 256];   // 32 KB
  const int bh   = blockIdx.x;
  const int lane = threadIdx.x & 31, wave = threadIdx.x >> 5;
  const int wm = wave >> 1, wn = wave & 1;
  const int m_base = wm * 16;      // d
  const int n_base = wn * 32;      // e
  const _Float16* A  = kT + (size_t)bh * DH_ * NPOS;
  const _Float16* Bt = vT + (size_t)bh * DH_ * NPOS;

  v8f acc[2] = {};
  for (int kp = 0; kp < NPOS; kp += 256) {
    if (wave == 0) {
      tdm_load_2d(lds_off_of(ldsA), A  + kp, 64, 256, NPOS);
      tdm_load_2d(lds_off_of(ldsB), Bt + kp, 64, 256, NPOS);
      __builtin_amdgcn_s_wait_tensorcnt(0);
    }
    __syncthreads();
    for (int k0 = 0; k0 < 256; k0 += 32) {
      v16h a0 = load_a_frag(ldsA, 256, m_base, k0, lane);
      v16h b0 = load_bt_frag(ldsB, 256, n_base,      k0, lane);
      v16h b1 = load_bt_frag(ldsB, 256, n_base + 16, k0, lane);
      acc[0] = wmma_f16(a0, b0, acc[0]);
      acc[1] = wmma_f16(a0, b1, acc[1]);
    }
    __syncthreads();    // WAR: panels reused next slab
  }
  _Float16* dst = ctxT + (size_t)bh * DH_ * DH_;
  const int m0 = m_base + ((lane >> 4) << 3);
  for (int tj = 0; tj < 2; ++tj) {
    const int e = n_base + tj * 16 + (lane & 15);
    for (int r = 0; r < 8; ++r)
      dst[e * DH_ + (m0 + r)] = (_Float16)acc[tj][r];
  }
}

// ---------------------------------------------------------------------------
// K7: out GEMM per (b,h): out[n][e] = (sum_d q[n][d]*ctx[d][e]) * d_inv[n]
// M=4096, K=64, N=64. ctx^T panel (64x64, 8KB) staged in LDS via TDM.
// Write outT[b][n][head*64+e] (f16) as B^T for K8.
// ---------------------------------------------------------------------------
__global__ void __launch_bounds__(256)
gemm_out_kernel(const _Float16* __restrict__ qh, const _Float16* __restrict__ ctxT,
                const float* __restrict__ dinv, _Float16* __restrict__ outT) {
  __shared__ _Float16 ldsB[64 * 64];   // 8 KB
  const int bh   = blockIdx.y;
  const int b    = bh >> 3, head = bh & 7;
  const int lane = threadIdx.x & 31, wave = threadIdx.x >> 5;
  const int wm = wave >> 1, wn = wave & 1;
  const int m_base = blockIdx.x * 128 + wm * 32;  // position n
  const int n_base = wn * 32;                     // e
  const _Float16* A  = qh   + (size_t)bh * NPOS * DH_;
  const float* dv = dinv + (size_t)bh * NPOS;

  if (wave == 0) {
    tdm_load_2d(lds_off_of(ldsB), ctxT + (size_t)bh * DH_ * DH_, 64, 64, 64);
    __builtin_amdgcn_s_wait_tensorcnt(0);
  }
  __syncthreads();

  v8f acc[2][2] = {};
  for (int k0 = 0; k0 < DH_; k0 += 32) {
    v16h a0 = load_a_frag(A, DH_, m_base,      k0, lane);
    v16h a1 = load_a_frag(A, DH_, m_base + 16, k0, lane);
    v16h b0 = load_bt_frag(ldsB, DH_, n_base,      k0, lane);
    v16h b1 = load_bt_frag(ldsB, DH_, n_base + 16, k0, lane);
    acc[0][0] = wmma_f16(a0, b0, acc[0][0]);
    acc[0][1] = wmma_f16(a0, b1, acc[0][1]);
    acc[1][0] = wmma_f16(a1, b0, acc[1][0]);
    acc[1][1] = wmma_f16(a1, b1, acc[1][1]);
  }
  for (int ti = 0; ti < 2; ++ti) {
    const int n0 = m_base + ti * 16 + ((lane >> 4) << 3);
    for (int tj = 0; tj < 2; ++tj) {
      const int e = n_base + tj * 16 + (lane & 15);
      for (int r = 0; r < 8; ++r) {
        const int n = n0 + r;
        float val = acc[ti][tj][r] * dv[n];
        outT[((size_t)b * NPOS + n) * HID_ + head * 64 + e] = (_Float16)val;
      }
    }
  }
}

// ---------------------------------------------------------------------------
// K8: final projection per batch: y[o][n] = sum_e Wo[o][e]*out[e][n] + bias[o]
// M=256, K=512, N=4096.  B panel staged as two 256-wide K slabs via TDM.
// Write f32 to d_out[b][o][n].
// ---------------------------------------------------------------------------
__global__ void __launch_bounds__(256)
gemm_final_kernel(const _Float16* __restrict__ Wo, const _Float16* __restrict__ outT,
                  const float* __restrict__ bias, float* __restrict__ y) {
  __shared__ _Float16 ldsB[64 * 256];   // 32 KB
  const int b    = blockIdx.z;
  const int lane = threadIdx.x & 31, wave = threadIdx.x >> 5;
  const int wm = wave >> 1, wn = wave & 1;
  const int m_base = blockIdx.y * 128 + wm * 32;  // out channel
  const int n_blk  = blockIdx.x * 64;
  const int n_base = n_blk + wn * 32;             // position
  const _Float16* Bt = outT + (size_t)b * NPOS * HID_;

  v8f acc[2][2] = {};
  for (int kp = 0; kp < HID_; kp += 256) {
    if (wave == 0) {
      tdm_load_2d(lds_off_of(ldsB), Bt + (size_t)n_blk * HID_ + kp, 64, 256, HID_);
      __builtin_amdgcn_s_wait_tensorcnt(0);
    }
    __syncthreads();
    for (int k0 = 0; k0 < 256; k0 += 32) {
      v16h a0 = load_a_frag(Wo, HID_, m_base,      kp + k0, lane);
      v16h a1 = load_a_frag(Wo, HID_, m_base + 16, kp + k0, lane);
      v16h b0 = load_bt_frag(ldsB, 256, wn * 32,      k0, lane);
      v16h b1 = load_bt_frag(ldsB, 256, wn * 32 + 16, k0, lane);
      acc[0][0] = wmma_f16(a0, b0, acc[0][0]);
      acc[0][1] = wmma_f16(a0, b1, acc[0][1]);
      acc[1][0] = wmma_f16(a1, b0, acc[1][0]);
      acc[1][1] = wmma_f16(a1, b1, acc[1][1]);
    }
    __syncthreads();    // WAR before next slab overwrite
  }
  for (int ti = 0; ti < 2; ++ti) {
    const int o0 = m_base + ti * 16 + ((lane >> 4) << 3);
    for (int tj = 0; tj < 2; ++tj) {
      const int n = n_base + tj * 16 + (lane & 15);
      for (int r = 0; r < 8; ++r) {
        const int o = o0 + r;
        y[((size_t)b * C_ + o) * NPOS + n] = acc[ti][tj][r] + bias[o];
      }
    }
  }
}

// ---------------------------------------------------------------------------
extern "C" void kernel_launch(void* const* d_in, const int* in_sizes, int n_in,
                              void* d_out, int out_size, void* d_ws, size_t ws_size,
                              hipStream_t stream) {
  const float* x     = (const float*)d_in[0];
  const float* gamma = (const float*)d_in[1];
  const float* beta  = (const float*)d_in[2];
  const float* w_qkv = (const float*)d_in[3];
  const float* w_out = (const float*)d_in[4];
  const float* b_out = (const float*)d_in[5];
  float* y = (float*)d_out;

  // Workspace carve-up (all offsets 256B-aligned); total ~154 MB.
  char* ws = (char*)d_ws;
  size_t cur = 0;
  auto carve = [&](size_t bytes) -> void* {
    void* p = ws + cur;
    cur += (bytes + 255) & ~(size_t)255;
    return p;
  };
  float*    mean   = (float*)carve(2048 * sizeof(float));
  float*    rstd   = (float*)carve(2048 * sizeof(float));
  _Float16* wqkv_h = (_Float16*)carve((size_t)OC3_ * C_ * 2);                 // 0.79 MB
  _Float16* wout_h = (_Float16*)carve((size_t)C_ * HID_ * 2);                 // 0.26 MB
  _Float16* xnT    = (_Float16*)carve((size_t)B_ * NPOS * C_ * 2);            // 16.8 MB
  _Float16* qh     = (_Float16*)carve((size_t)B_ * HEADS_ * NPOS * DH_ * 2);  // 33.6 MB
  _Float16* kT     = (_Float16*)carve((size_t)B_ * HEADS_ * DH_ * NPOS * 2);  // 33.6 MB
  _Float16* vT     = (_Float16*)carve((size_t)B_ * HEADS_ * DH_ * NPOS * 2);  // 33.6 MB
  float*    ksum   = (float*)carve((size_t)B_ * HEADS_ * DH_ * sizeof(float));
  float*    dinv   = (float*)carve((size_t)B_ * HEADS_ * NPOS * sizeof(float));
  _Float16* ctxT   = (_Float16*)carve((size_t)B_ * HEADS_ * DH_ * DH_ * 2);
  _Float16* outT   = (_Float16*)carve((size_t)B_ * NPOS * HID_ * 2);          // 33.6 MB
  (void)ws_size; (void)in_sizes; (void)n_in; (void)out_size;

  stats_kernel<<<2048, 256, 0, stream>>>(x, mean, rstd);
  norm_transpose_kernel<<<2048, 256, 0, stream>>>(x, gamma, beta, mean, rstd, xnT);
  f32_to_f16_kernel<<<(OC3_ * C_) / 256, 256, 0, stream>>>(w_qkv, wqkv_h, OC3_ * C_);
  f32_to_f16_kernel<<<(C_ * HID_) / 256, 256, 0, stream>>>(w_out, wout_h, C_ * HID_);

  gemm_qkv_kernel<<<dim3(NPOS / 64, OC3_ / 128, B_), 256, 0, stream>>>(wqkv_h, xnT, qh, kT, vT);

  ksum_kernel<<<B_ * HEADS_ * DH_, 256, 0, stream>>>(kT, ksum);
  gemm_ctx_kernel<<<B_ * HEADS_, 256, 0, stream>>>(kT, vT, ctxT);
  dinv_kernel<<<(B_ * HEADS_ * NPOS) / 256, 256, 0, stream>>>(qh, ksum, dinv);

  gemm_out_kernel<<<dim3(NPOS / 128, B_ * HEADS_), 256, 0, stream>>>(qh, ctxT, dinv, outT);
  gemm_final_kernel<<<dim3(NPOS / 64, C_ / 128, B_), 256, 0, stream>>>(wout_h, outT, b_out, y);
}
